// Quantize_4200478015514
// MI455X (gfx1250) — matrix operand order repaired and checked by reading, not compiled, not run on hardware
//
#include <hip/hip_runtime.h>
#include <hip/hip_bf16.h>

// ---------------- problem constants ----------------
#define TT 2048
#define BB 16
#define DIM 256
#define NEMB 1024
#define NROWS (TT*BB)            // 32768
#define QELEMS (NROWS*DIM)       // 8388608

// ---------------- tiling ----------------
#define TPB 256                  // 8 waves of 32
#define MB 128                   // rows per workgroup
#define NCHUNK 64                // codebook columns staged per chunk
#define NCHUNKS (NEMB / NCHUNK)  // 16
#define STR 264                  // padded bf16 stride (264*2B = 33*16B -> 4-bank skew, 16B aligned)

// ---------------- CDNA5 async-copy availability ----------------
#if defined(__AMDGCN__) && \
    __has_builtin(__builtin_amdgcn_global_load_async_to_lds_b128) && \
    __has_builtin(__builtin_amdgcn_s_wait_asynccnt)
#define USE_ASYNC 1
#define NBUF 2
#else
#define USE_ASYNC 0
#define NBUF 1
#endif

#define AS3 __attribute__((address_space(3)))
#define AS1 __attribute__((address_space(1)))

typedef int v4i __attribute__((vector_size(16)));   // matches builtin's int4 param type

__device__ __forceinline__ AS1 v4i* gcast(const void* p) {
    // global flat address == addrspace(1) address
    return (AS1 v4i*)(unsigned long long)p;
}
__device__ __forceinline__ AS3 v4i* lcast(void* p) {
    // flat LDS address: low 32 bits are the workgroup-relative LDS offset
    return (AS3 v4i*)(unsigned)(unsigned long long)p;
}

typedef __attribute__((ext_vector_type(16))) __bf16 v16bf;
typedef __attribute__((ext_vector_type(8)))  float  v8f;

union Frag16 { v16bf v; uint4 u[2]; };

__device__ __forceinline__ unsigned short f2bf(float f) {
    unsigned u = __float_as_uint(f);
    u += 0x7FFFu + ((u >> 16) & 1u);   // round-to-nearest-even
    return (unsigned short)(u >> 16);
}

// ---------------- kernel 1: code norms + bf16 transposed codebook + zero accumulators ----
__global__ void vq_init(const float* __restrict__ embed,
                        float* __restrict__ enorm,
                        unsigned short* __restrict__ ebf,   // [NEMB][DIM] bf16, code-major
                        unsigned* __restrict__ counts,
                        unsigned* __restrict__ msum,
                        float* __restrict__ diffsum) {
    int n = blockIdx.x * blockDim.x + threadIdx.x;
    if (n < NEMB) {
        float s = 0.0f;
        for (int k = 0; k < DIM; ++k) {
            float e = embed[k * NEMB + n];
            s += e * e;
            ebf[n * DIM + k] = f2bf(e);
        }
        enorm[n]  = s;
        counts[n] = 0u;
    }
    if (n == 0) { *msum = 0u; *diffsum = 0.0f; }
}

// ---------------- kernel 2: WMMA distance + argmin + gather ----------------
__global__ __launch_bounds__(TPB) void vq_main(
        const float* __restrict__ inp,           // [NROWS, DIM]
        const unsigned char* __restrict__ mask,  // [NROWS] bool
        const float* __restrict__ embed,         // [DIM, NEMB]
        const float* __restrict__ enorm,         // [NEMB]
        const unsigned short* __restrict__ ebf,  // [NEMB, DIM] bf16
        float* __restrict__ outq,                // [NROWS, DIM]
        float* __restrict__ outind,              // [NROWS] (as float)
        unsigned* __restrict__ counts,
        unsigned* __restrict__ msum,
        float* __restrict__ diffsum) {
    __shared__ unsigned short ldsA[MB * STR];            // 67584 B: bf16 input rows
    __shared__ unsigned short ldsB[NBUF][NCHUNK * STR];  // 33792 B per buf: bf16 codes [n][k]

    const int tid  = threadIdx.x;
    const int lane = tid & 31;
    const int w    = tid >> 5;        // wave id 0..7
    const int hi   = lane >> 4;       // half-wave select
    const int ln   = lane & 15;
    const int row0 = blockIdx.x * MB;

    // stage A: fp32 -> bf16 rows for this block
    for (int e = tid; e < MB * DIM; e += TPB) {
        int r = e >> 8, k = e & (DIM - 1);
#if USE_ASYNC
        __builtin_prefetch(&inp[(row0 + r) * DIM + k + 2 * TPB], 0, 0);
#endif
        ldsA[r * STR + k] = f2bf(inp[(row0 + r) * DIM + k]);
    }

#if USE_ASYNC
    // issue async copy of chunk 0 (bf16 codebook rows -> padded LDS tile); does not touch ldsA
    {
#pragma unroll
        for (int j = 0; j < 8; ++j) {
            int t  = tid + j * TPB;         // 2048 x 16B transfers per chunk
            int n  = t >> 5;
            int kb = t & 31;
            __builtin_amdgcn_global_load_async_to_lds_b128(
                gcast(ebf + (unsigned)n * DIM + kb * 8),
                lcast(&ldsB[0][n * STR + kb * 8]), 0, 0);
        }
    }
#endif

    __syncthreads();   // A staged and visible to all waves

    // registerize this wave's whole A tile (16 rows x 256 K bf16 = 8 fragments, chunk-invariant)
    const unsigned short* Ap = &ldsA[(w * 16 + ln) * STR];  // lane ln holds row M=ln
    Frag16 afr[8];
#pragma unroll
    for (int kb = 0; kb < 8; ++kb) {
        // A 16x32 bf16: lanes<16 get K 0..15, lanes>=16 get K 16..31
        afr[kb].u[0] = *(const uint4*)(Ap + kb * 32 + hi * 16);
        afr[kb].u[1] = *(const uint4*)(Ap + kb * 32 + hi * 16 + 8);
    }

    float bestv[8];
    int   bestn[8];
#pragma unroll
    for (int r = 0; r < 8; ++r) { bestv[r] = 3.4e38f; bestn[r] = 0; }

    for (int c = 0; c < NCHUNKS; ++c) {
        const int n0 = c * NCHUNK;
#if USE_ASYNC
        __builtin_amdgcn_s_wait_asynccnt(0);
        __syncthreads();                  // chunk c visible to all waves
        const int buf = c & 1;
        if (c + 1 < NCHUNKS) {            // prefetch chunk c+1 into the other buffer
            const int p0 = n0 + NCHUNK;
#pragma unroll
            for (int j = 0; j < 8; ++j) {
                int t  = tid + j * TPB;
                int n  = t >> 5;
                int kb = t & 31;
                __builtin_amdgcn_global_load_async_to_lds_b128(
                    gcast(ebf + (unsigned)(p0 + n) * DIM + kb * 8),
                    lcast(&ldsB[buf ^ 1][n * STR + kb * 8]), 0, 0);
            }
        }
#else
        const int buf = 0;
        __syncthreads();                  // protect ldsB reuse
        for (int e = tid; e < NCHUNK * DIM; e += TPB) {
            int k = e >> 6, n = e & (NCHUNK - 1);
            ldsB[0][n * STR + k] = f2bf(embed[k * NEMB + n0 + n]);
        }
        __syncthreads();
#endif

        // prefetch this chunk's per-lane code norms so their latency hides under the WMMAs
        float enq[4];
#pragma unroll
        for (int t = 0; t < 4; ++t) enq[t] = enorm[n0 + t * 16 + ln];

        // 4 N-tiles of 16 codes each — unrolled: B ds_loads use immediate offsets off one base
        const unsigned short* Bbase = &ldsB[buf][ln * STR + hi * 16];
#pragma unroll
        for (int t = 0; t < 4; ++t) {
            const unsigned short* Bp = Bbase + t * 16 * STR; // lane ln holds column N=ln
            Frag16 b;
            v8f acc = {};
#pragma unroll
            for (int kb = 0; kb < 8; ++kb) {
                const int k0 = kb * 32;
                // B 32x16 bf16: lanes<16 get K 0..15, lanes>=16 get K 16..31
                b.u[0] = *(const uint4*)(Bp + k0);
                b.u[1] = *(const uint4*)(Bp + k0 + 8);
                acc = __builtin_amdgcn_wmma_f32_16x16x32_bf16(
                        false, afr[kb].v, false, b.v, (short)0, acc, false, false);
            }
            // dist (per row, up to a row-constant) = ||e_n||^2 - 2*score
            const int n = n0 + t * 16 + ln;
#pragma unroll
            for (int r = 0; r < 8; ++r) {
                float v = enq[t] - 2.0f * acc[r];
                if (v < bestv[r]) { bestv[r] = v; bestn[r] = n; }
            }
        }
    }

    // argmin across the 16 lanes holding one row (xor<16 stays within half-wave)
#pragma unroll
    for (int r = 0; r < 8; ++r) {
#pragma unroll
        for (int off = 1; off < 16; off <<= 1) {
            float ov = __shfl_xor(bestv[r], off, 32);
            int   on = __shfl_xor(bestn[r], off, 32);
            if (ov < bestv[r] || (ov == bestv[r] && on < bestn[r])) {
                bestv[r] = ov; bestn[r] = on;
            }
        }
    }

    // lanes 0 / 16 own rows r / r+8: write indices + assignment stats
    if (ln == 0) {
#pragma unroll
        for (int r = 0; r < 8; ++r) {
            int grow = row0 + w * 16 + r + 8 * hi;
            outind[grow] = (float)bestn[r];
            if (mask[grow]) {
                atomicAdd(&counts[bestn[r]], 1u);
                atomicAdd(msum, 1u);
            }
        }
    }

    // gather winning codes (exact fp32 codebook), mask, accumulate diff
    float dsum = 0.0f;
#pragma unroll
    for (int m = 0; m < 16; ++m) {
        int idx  = __shfl(bestn[m & 7], (m >> 3) * 16, 32);
        int grow = row0 + w * 16 + m;
        float mk = mask[grow] ? 1.0f : 0.0f;
        for (int j = lane; j < DIM; j += 32) {
            float q = embed[j * NEMB + idx] * mk;
            float x = inp[grow * DIM + j] * mk;
            outq[grow * DIM + j] = q;   // straight-through forward value == masked quantize
            float d = q - x;
            dsum += d * d;
        }
    }
#pragma unroll
    for (int off = 1; off < 32; off <<= 1) dsum += __shfl_xor(dsum, off, 32);
    if (lane == 0) atomicAdd(diffsum, dsum);
}

// ---------------- kernel 3: scalars ----------------
__global__ void vq_final(const unsigned* __restrict__ counts,
                         const unsigned* __restrict__ msum,
                         const float* __restrict__ diffsum,
                         float* __restrict__ out_diff,
                         float* __restrict__ out_eff) {
    __shared__ float red[256];
    float ms = (float)(*msum);
    float s = 0.0f;
    for (int n = threadIdx.x; n < NEMB; n += 256) {
        float f = (float)counts[n] / ms;
        s += f * f;
    }
    red[threadIdx.x] = s;
    __syncthreads();
    for (int st = 128; st > 0; st >>= 1) {
        if (threadIdx.x < (unsigned)st) red[threadIdx.x] += red[threadIdx.x + st];
        __syncthreads();
    }
    if (threadIdx.x == 0) {
        *out_eff  = 1.0f / red[0];
        *out_diff = (*diffsum) / (float)QELEMS;
    }
}

extern "C" void kernel_launch(void* const* d_in, const int* in_sizes, int n_in,
                              void* d_out, int out_size, void* d_ws, size_t ws_size,
                              hipStream_t stream) {
    const float*         inp   = (const float*)d_in[0];
    const unsigned char* mask  = (const unsigned char*)d_in[1];
    const float*         embed = (const float*)d_in[2];

    float* out     = (float*)d_out;
    float* outq    = out;                 // [0, 8388608)
    float* outdiff = out + QELEMS;        // [8388608]
    float* outind  = out + QELEMS + 1;    // [8388609, 8421377)
    float* outeff  = out + QELEMS + 1 + NROWS;  // [8421377]

    float*          enorm   = (float*)d_ws;                       // 1024 f32
    unsigned*       counts  = (unsigned*)((char*)d_ws + 4096);    // 1024 u32
    unsigned*       msum    = (unsigned*)((char*)d_ws + 8192);    // 1 u32
    float*          diffsum = (float*)((char*)d_ws + 8196);       // 1 f32
    unsigned short* ebf     = (unsigned short*)((char*)d_ws + 16384); // 1024*256 bf16 (512KB)

    vq_init<<<NEMB / 256, 256, 0, stream>>>(embed, enorm, ebf, counts, msum, diffsum);
    vq_main<<<NROWS / MB, TPB, 0, stream>>>(inp, mask, embed, enorm, ebf,
                                            outq, outind, counts, msum, diffsum);
    vq_final<<<1, 256, 0, stream>>>(counts, msum, diffsum, outdiff, outeff);
}
